// LowDimQKMultiHeadAttention_54142357733649
// MI455X (gfx1250) — matrix operand order repaired and checked by reading, not compiled, not run on hardware
//
#include <hip/hip_runtime.h>
#include <math.h>

typedef __attribute__((ext_vector_type(16))) _Float16 v16h;
typedef __attribute__((ext_vector_type(8)))  float    v8f;

#define BATCH 4
#define TQ 2048
#define TK 2048
#define DIN 1024
#define NH 8
#define HQK 32
#define HV 128
#define QKD 256

// ---- CDNA5 async global->LDS DMA (ASYNCcnt), guarded by __has_builtin ------
#if defined(__has_builtin)
#if __has_builtin(__builtin_amdgcn_global_load_async_to_lds_b128) && \
    __has_builtin(__builtin_amdgcn_global_load_async_to_lds_b64)
#define USE_ASYNC 1
#endif
#endif
#ifndef USE_ASYNC
#define USE_ASYNC 0
#endif

#if USE_ASYNC
// Exact parameter types per hipcc diagnostic:
//   'int __attribute__((vector_size(N*sizeof(int)))) __device__ *'  (addrspace(1))
typedef int i32x2 __attribute__((vector_size(8)));
typedef int i32x4 __attribute__((vector_size(16)));
typedef __attribute__((address_space(1))) i32x2* g2p;
typedef __attribute__((address_space(1))) i32x4* g4p;
typedef __attribute__((address_space(3))) i32x2* l2p;
typedef __attribute__((address_space(3))) i32x4* l4p;
#endif

__device__ __forceinline__ void wait_async_zero() {
#if defined(__has_builtin) && __has_builtin(__builtin_amdgcn_s_wait_asynccnt)
  __builtin_amdgcn_s_wait_asynccnt(0);
#else
  asm volatile("s_wait_asynccnt 0" ::: "memory");
#endif
}

union Frag { unsigned u[8]; v16h v; };

__device__ __forceinline__ v8f wmma16x16x32(const Frag& a, const Frag& b, v8f c) {
  // D = A(16x32 f16) * B(32x16 f16) + C(16x16 f32)
  return __builtin_amdgcn_wmma_f32_16x16x32_f16(false, a.v, false, b.v,
                                                (short)0, c, false, false);
}

// ---------------------------------------------------------------------------
// Projection: Y[M x 256] = f16( (X[M x 1024] @ W[1024 x 256] + bias) * scale )
// ---------------------------------------------------------------------------
#define PBM 128
#define PBN 64
#define PBK 32
#define PKP (PBK + 8)

__global__ __launch_bounds__(256)
void proj_kernel(const float* __restrict__ X, const float* __restrict__ W,
                 const float* __restrict__ bias, _Float16* __restrict__ Y,
                 float out_scale) {
  __shared__ _Float16 As[PBM][PKP];
  __shared__ _Float16 Bs[PBN][PKP];   // W tile transposed: Bs[n][k]

  const int tid  = threadIdx.x;
  const int lane = tid & 31;
  const int wave = tid >> 5;
  const int hi   = (lane >> 4) & 1;
  const int m16  = lane & 15;

  const int mtile = blockIdx.x >> 2;
  const int ntile = blockIdx.x & 3;
  const int row0  = mtile * PBM;
  const int col0  = ntile * PBN;

  const int wr = wave >> 1;
  const int wc = wave & 1;

  v8f acc[2][2];
  #pragma unroll
  for (int i = 0; i < 2; ++i)
    #pragma unroll
    for (int j = 0; j < 2; ++j)
      #pragma unroll
      for (int r = 0; r < 8; ++r) acc[i][j][r] = 0.f;

  for (int kb = 0; kb < DIN; kb += PBK) {
    #pragma unroll
    for (int i = 0; i < 4; ++i) {
      int lin = i * 1024 + tid * 4;
      int r = lin >> 5, c = lin & 31;
      float4 a4 = *reinterpret_cast<const float4*>(
          X + (size_t)(row0 + r) * DIN + kb + c);
      As[r][c+0] = (_Float16)a4.x; As[r][c+1] = (_Float16)a4.y;
      As[r][c+2] = (_Float16)a4.z; As[r][c+3] = (_Float16)a4.w;
    }
    #pragma unroll
    for (int i = 0; i < 2; ++i) {
      int lin = i * 1024 + tid * 4;
      int kr = lin >> 6, nc = lin & 63;
      float4 w4 = *reinterpret_cast<const float4*>(
          W + (size_t)(kb + kr) * QKD + col0 + nc);
      Bs[nc+0][kr] = (_Float16)w4.x; Bs[nc+1][kr] = (_Float16)w4.y;
      Bs[nc+2][kr] = (_Float16)w4.z; Bs[nc+3][kr] = (_Float16)w4.w;
    }
    __syncthreads();

    Frag afrag[2], bfrag[2];
    #pragma unroll
    for (int ar = 0; ar < 2; ++ar) {
      const unsigned* ap = reinterpret_cast<const unsigned*>(
          &As[wr*32 + ar*16 + m16][0]);
      #pragma unroll
      for (int j = 0; j < 8; ++j) {
        int dj = ((j < 4) ? j : (j + 4)) + (hi ? 4 : 0);
        afrag[ar].u[j] = ap[dj];
      }
    }
    #pragma unroll
    for (int bc = 0; bc < 2; ++bc) {
      const unsigned* bp = reinterpret_cast<const unsigned*>(
          &Bs[wc*32 + bc*16 + m16][0]);
      #pragma unroll
      for (int j = 0; j < 8; ++j)
        bfrag[bc].u[j] = bp[((j < 4) ? j : (j + 4)) + (hi ? 4 : 0)];
    }
    #pragma unroll
    for (int ar = 0; ar < 2; ++ar)
      #pragma unroll
      for (int bc = 0; bc < 2; ++bc)
        acc[ar][bc] = wmma16x16x32(afrag[ar], bfrag[bc], acc[ar][bc]);
    __syncthreads();
  }

  #pragma unroll
  for (int ar = 0; ar < 2; ++ar)
    #pragma unroll
    for (int bc = 0; bc < 2; ++bc) {
      int colN = col0 + wc*32 + bc*16 + m16;
      float bv = bias[colN];
      #pragma unroll
      for (int r = 0; r < 8; ++r) {
        int rowG = row0 + wr*32 + ar*16 + r + (hi ? 8 : 0);
        Y[(size_t)rowG * QKD + colN] = (_Float16)((acc[ar][bc][r] + bv) * out_scale);
      }
    }
}

// ---------------------------------------------------------------------------
// V transpose+convert: Vt[b][h][d][k] (f16, dim-major) from V[b][k][h*128+d]
// ---------------------------------------------------------------------------
#define VCP 80   // padded transpose row (halves); 160B stride, 16B aligned

__global__ __launch_bounds__(256)
void vconv_kernel(const float* __restrict__ V, _Float16* __restrict__ vt) {
  __shared__ _Float16 T[HV][VCP];
  const int tid = threadIdx.x;
  const int kt  = blockIdx.x & 31;          // key tile of 64
  const int h   = (blockIdx.x >> 5) & 7;
  const int b   = blockIdx.x >> 8;

  #pragma unroll
  for (int i = 0; i < 8; ++i) {
    int lin = i * 1024 + tid * 4;
    int k = lin >> 7;        // 0..63
    int d = lin & 127;
    float4 v4 = *reinterpret_cast<const float4*>(
        V + ((size_t)(b * TK + kt * 64 + k)) * DIN + h * HV + d);
    T[d+0][k] = (_Float16)v4.x; T[d+1][k] = (_Float16)v4.y;
    T[d+2][k] = (_Float16)v4.z; T[d+3][k] = (_Float16)v4.w;
  }
  __syncthreads();

  int d = tid >> 1;
  int p = (tid & 1) * 32;
  const uint4* src = reinterpret_cast<const uint4*>(&T[d][p]);
  uint4* dst = reinterpret_cast<uint4*>(
      vt + (((size_t)(b * NH + h)) * HV + d) * TK + kt * 64 + p);
  #pragma unroll
  for (int i = 0; i < 4; ++i) dst[i] = src[i];
}

// ---------------------------------------------------------------------------
// Flash attention, double-buffered LDS staging (async DMA when available).
// ---------------------------------------------------------------------------
#define AKP (HQK + 8)   // 40 halves, 80B row
#define VKP (32 + 8)    // 40 halves, 80B row

__global__ __launch_bounds__(256)
void attn_kernel(const _Float16* __restrict__ qh, const _Float16* __restrict__ kh,
                 const _Float16* __restrict__ vt, const unsigned char* __restrict__ mask,
                 float* __restrict__ out) {
  __shared__ _Float16 Ks[2][32][AKP];
  __shared__ _Float16 Vs[2][HV][VKP];
  __shared__ _Float16 Ps[8][16][AKP];

  const int tid  = threadIdx.x;
  const int lane = tid & 31;
  const int wave = tid >> 5;
  const int hi   = (lane >> 4) & 1;
  const int m16  = lane & 15;

  const int qt = blockIdx.x & 15;
  const int h  = (blockIdx.x >> 4) & 7;
  const int b  = blockIdx.x >> 7;

  const int qrow0 = qt * 128 + wave * 16;

  // staging assignments (fixed per thread)
  const int skr = tid >> 3;                 // Ks key row 0..31
  const int skc = (tid & 7) * 4;            // Ks dim col (halves)
  const int svd = tid >> 1;                 // Vs dim row 0..127
  const int svp = (tid & 1) * 16;           // Vs key part (halves)

  const _Float16* kbase = kh + ((size_t)(b * TK + skr)) * QKD + h * HQK + skc;
  const _Float16* vbase = vt + (((size_t)(b * NH + h)) * HV + svd) * TK + svp;

  // q A-fragment (16x32); 1/sqrt(head_qk) folded in during projection
  Frag qf;
  {
    const unsigned* qp = reinterpret_cast<const unsigned*>(
        qh + ((size_t)(b * TQ + qrow0 + m16)) * QKD + h * HQK);
    #pragma unroll
    for (int j = 0; j < 8; ++j) {
      int dj = ((j < 4) ? j : (j + 4)) + (hi ? 4 : 0);
      qf.u[j] = qp[dj];
    }
  }

  v8f acc[8];
  float mi[8], li[8];
  #pragma unroll
  for (int c = 0; c < 8; ++c)
    #pragma unroll
    for (int r = 0; r < 8; ++r) acc[c][r] = 0.f;
  #pragma unroll
  for (int r = 0; r < 8; ++r) { mi[r] = -__builtin_inff(); li[r] = 0.f; }

  const v8f zacc = {0.f,0.f,0.f,0.f,0.f,0.f,0.f,0.f};

  // stage chunk (32 keys) into LDS buffer `buf`
  auto stage = [&](int buf, int kb) {
    _Float16* ksrc = const_cast<_Float16*>(kbase + (size_t)kb * QKD);
    _Float16* vsrc = const_cast<_Float16*>(vbase + kb);
    _Float16* kdst = &Ks[buf][skr][skc];
    _Float16* vdst = &Vs[buf][svd][svp];
#if USE_ASYNC
    __builtin_amdgcn_global_load_async_to_lds_b64(
        (g2p)(void*)ksrc, (l2p)(void*)kdst, 0, 0);
    __builtin_amdgcn_global_load_async_to_lds_b128(
        (g4p)(void*)vsrc, (l4p)(void*)vdst, 0, 0);
#else
    *reinterpret_cast<uint2*>(kdst) = *reinterpret_cast<const uint2*>(ksrc);
    *reinterpret_cast<uint4*>(vdst) = *reinterpret_cast<const uint4*>(vsrc);
#endif
  };

  stage(0, 0);
#if USE_ASYNC
  wait_async_zero();
#endif
  __syncthreads();

  for (int kc = 0; kc < TK / 32; ++kc) {
    const int cur = kc & 1;
    const int kb  = kc * 32;
    if (kc + 1 < TK / 32) stage(cur ^ 1, kb + 32);   // overlap next-chunk DMA

    // scores S = q * k^T : two 16x16 tiles
    v8f sc[2];
    #pragma unroll
    for (int t = 0; t < 2; ++t) {
      Frag bf;
      const unsigned* bp =
          reinterpret_cast<const unsigned*>(&Ks[cur][t*16 + m16][0]);
      #pragma unroll
      for (int j = 0; j < 8; ++j) bf.u[j] = bp[((j < 4) ? j : (j + 4)) + (hi ? 4 : 0)];
      sc[t] = wmma16x16x32(qf, bf, zacc);
      int kg = kb + t*16 + m16;
      bool msk = mask[(size_t)b * TK + kg] != 0;
      #pragma unroll
      for (int r = 0; r < 8; ++r) {
        float v = sc[t][r];
        if (msk || v != v) v = -__builtin_inff();    // mask + NaN -> -inf
        sc[t][r] = v;
      }
    }

    // online softmax; row M = r + 8*hi lives in one 16-lane half of one VGPR
    float p0[8], p1[8], alpha[8];
    #pragma unroll
    for (int r = 0; r < 8; ++r) {
      float rmax = fmaxf(sc[0][r], sc[1][r]);
      #pragma unroll
      for (int off = 1; off < 16; off <<= 1)
        rmax = fmaxf(rmax, __shfl_xor(rmax, off, 32));
      float mnew = fmaxf(mi[r], rmax);
      float da = mi[r] - mnew;
      alpha[r] = (da == da) ? __expf(da) : 1.0f;
      float e0 = sc[0][r] - mnew, e1 = sc[1][r] - mnew;
      p0[r] = (e0 == e0) ? __expf(e0) : 0.f;
      p1[r] = (e1 == e1) ? __expf(e1) : 0.f;
      float rs = p0[r] + p1[r];
      #pragma unroll
      for (int off = 1; off < 16; off <<= 1)
        rs += __shfl_xor(rs, off, 32);
      li[r] = li[r] * alpha[r] + rs;
      mi[r] = mnew;
    }
    #pragma unroll
    for (int c = 0; c < 8; ++c)
      #pragma unroll
      for (int r = 0; r < 8; ++r) acc[c][r] *= alpha[r];

    // restage P: C-layout -> row-major 16x32 (same-wave DS ops are in-order)
    #pragma unroll
    for (int r = 0; r < 8; ++r) {
      int row = r + (hi ? 8 : 0);
      Ps[wave][row][m16]      = (_Float16)p0[r];
      Ps[wave][row][16 + m16] = (_Float16)p1[r];
    }
    Frag pf;
    {
      const unsigned* pp =
          reinterpret_cast<const unsigned*>(&Ps[wave][m16][0]);
      #pragma unroll
      for (int j = 0; j < 8; ++j) {
        int dj = ((j < 4) ? j : (j + 4)) + (hi ? 4 : 0);
        pf.u[j] = pp[dj];
      }
    }

    // O += P(16x32) * V(32x128): 8 column tiles
    #pragma unroll
    for (int c = 0; c < 8; ++c) {
      Frag vf;
      const unsigned* vp =
          reinterpret_cast<const unsigned*>(&Vs[cur][c*16 + m16][0]);
      #pragma unroll
      for (int j = 0; j < 8; ++j) vf.u[j] = vp[((j < 4) ? j : (j + 4)) + (hi ? 4 : 0)];
      acc[c] = wmma16x16x32(pf, vf, acc[c]);
    }

#if USE_ASYNC
    wait_async_zero();     // next chunk's DMA must land before the barrier
#endif
    __syncthreads();
  }

  float invl[8];
  #pragma unroll
  for (int r = 0; r < 8; ++r) invl[r] = (li[r] > 0.f) ? (1.0f / li[r]) : 0.f;
  #pragma unroll
  for (int c = 0; c < 8; ++c)
    #pragma unroll
    for (int r = 0; r < 8; ++r) {
      int row = qrow0 + r + (hi ? 8 : 0);
      out[((size_t)(b * TQ + row)) * DIN + h * HV + c*16 + m16] =
          acc[c][r] * invl[r];
    }
}

// ---------------------------------------------------------------------------
extern "C" void kernel_launch(void* const* d_in, const int* in_sizes, int n_in,
                              void* d_out, int out_size, void* d_ws, size_t ws_size,
                              hipStream_t stream) {
  const float* Q  = (const float*)d_in[0];
  const float* K  = (const float*)d_in[1];
  const float* V  = (const float*)d_in[2];
  const float* Wq = (const float*)d_in[3];
  const float* bq = (const float*)d_in[4];
  const float* Wk = (const float*)d_in[5];
  const float* bk = (const float*)d_in[6];
  const unsigned char* mask = (const unsigned char*)d_in[7];
  float* out = (float*)d_out;

  _Float16* qws = (_Float16*)d_ws;                          // 4 MB
  _Float16* kws = qws + (size_t)BATCH * TQ * QKD;           // 4 MB
  _Float16* vts = kws + (size_t)BATCH * TK * QKD;           // 16 MB

  const float scale = 1.0f / sqrtf((float)HQK);

  proj_kernel<<<dim3(256), dim3(256), 0, stream>>>(Q, Wq, bq, qws, scale);
  proj_kernel<<<dim3(256), dim3(256), 0, stream>>>(K, Wk, bk, kws, 1.0f);
  vconv_kernel<<<dim3(1024), dim3(256), 0, stream>>>(V, vts);
  attn_kernel<<<dim3(512), dim3(256), 0, stream>>>(qws, kws, vts, mask, out);
}
